// Diversity6_52699248721938
// MI455X (gfx1250) — matrix operand order repaired and checked by reading, not compiled, not run on hardware
//
#include <hip/hip_runtime.h>
#include <hip/hip_bf16.h>

#define NB      4096      // batch
#define NC      1000      // classes
#define KPAD    1024      // K padded to multiple of 32 for WMMA
#define NM      6         // models
#define INV_T   (1.0f / 20.0f)
#define SCALE_F 0.3f

typedef _Float16 v8h  __attribute__((ext_vector_type(8)));
typedef _Float16 v16h __attribute__((ext_vector_type(16)));
typedef float    v8f  __attribute__((ext_vector_type(8)));

__device__ __forceinline__ float wred_sum(float v) {
#pragma unroll
    for (int o = 16; o > 0; o >>= 1) v += __shfl_xor(v, o, 32);
    return v;
}
__device__ __forceinline__ float wred_max(float v) {
#pragma unroll
    for (int o = 16; o > 0; o >>= 1) v = fmaxf(v, __shfl_xor(v, o, 32));
    return v;
}

// Block-wide reductions over 256 threads (8 wave32s). All threads return the result.
__device__ __forceinline__ float bred_sum(float v, float* red) {
    const int wid = threadIdx.x >> 5;
    v = wred_sum(v);
    __syncthreads();                       // protect red[] reuse
    if ((threadIdx.x & 31) == 0) red[wid] = v;
    __syncthreads();
    float t = 0.0f;
#pragma unroll
    for (int i = 0; i < 8; ++i) t += red[i];
    return t;
}
__device__ __forceinline__ float bred_max(float v, float* red) {
    const int wid = threadIdx.x >> 5;
    v = wred_max(v);
    __syncthreads();
    if ((threadIdx.x & 31) == 0) red[wid] = v;
    __syncthreads();
    float t = red[0];
#pragma unroll
    for (int i = 1; i < 8; ++i) t = fmaxf(t, red[i]);
    return t;
}

// One block (256 threads) per sample b.
// Stage 6x1000 f32 rows -> LDS; per model: row max, exp, sum(e), sum(e^2);
// write u_m = (e/Z - 1/C) * rsqrt(sum(p^2) - 1/C) to f16 LDS tile;
// wave 0 computes the 16x16 sample Gram U*U^T with v_wmma_f32_16x16x32_f16
// and sums the strict upper 6x6 triangle.
__global__ __launch_bounds__(256) void div6_main(
    const float* __restrict__ o0, const float* __restrict__ o1,
    const float* __restrict__ o2, const float* __restrict__ o3,
    const float* __restrict__ o4, const float* __restrict__ o5,
    float* __restrict__ partial)
{
    __shared__ float    xbuf[NM * NC];      // 24000 B staging (f32 logits)
    __shared__ _Float16 ubuf[NM][KPAD];     // 12288 B f16 U tile (rows 0..5)
    __shared__ float    red[8];

    const int   tid  = threadIdx.x;
    const int   b    = blockIdx.x;
    const float invC = 1.0f / (float)NC;

    // ---- stage: 250 float4 per row, 16B-aligned (1000 floats, b*4000B offset) ----
    {
        const size_t base = (size_t)b * NC;
        if (tid < 250) {
            ((float4*)xbuf)[0 * 250 + tid] = ((const float4*)(o0 + base))[tid];
            ((float4*)xbuf)[1 * 250 + tid] = ((const float4*)(o1 + base))[tid];
            ((float4*)xbuf)[2 * 250 + tid] = ((const float4*)(o2 + base))[tid];
            ((float4*)xbuf)[3 * 250 + tid] = ((const float4*)(o3 + base))[tid];
            ((float4*)xbuf)[4 * 250 + tid] = ((const float4*)(o4 + base))[tid];
            ((float4*)xbuf)[5 * 250 + tid] = ((const float4*)(o5 + base))[tid];
        }
    }
    __syncthreads();

    // ---- per-model softmax stats + u write (each c owned by exactly one thread) ----
#pragma unroll
    for (int m = 0; m < NM; ++m) {
        const float* xr = &xbuf[m * NC];
        float a0 = xr[tid];
        float a1 = xr[tid + 256];
        float a2 = xr[tid + 512];
        float a3 = (tid < 232) ? xr[tid + 768] : -3.0e38f;

        float mx = bred_max(fmaxf(fmaxf(a0, a1), fmaxf(a2, a3)), red);

        float e0 = __expf((a0 - mx) * INV_T);
        float e1 = __expf((a1 - mx) * INV_T);
        float e2 = __expf((a2 - mx) * INV_T);
        float e3 = (tid < 232) ? __expf((a3 - mx) * INV_T) : 0.0f;

        float Z = bred_sum(e0 + e1 + e2 + e3, red);
        float Q = bred_sum(e0 * e0 + e1 * e1 + e2 * e2 + e3 * e3, red);

        // p = e/Z ; var = sum(p^2) - 1/C ; u = (p - 1/C) * rsqrt(var)
        const float rz  = 1.0f / Z;
        const float var = fmaxf(Q * rz * rz - invC, 1.0e-30f);
        const float rn  = rsqrtf(var);
        const float am  = rn * rz;
        const float bm  = rn * invC;

        ubuf[m][tid]       = (_Float16)(e0 * am - bm);
        ubuf[m][tid + 256] = (_Float16)(e1 * am - bm);
        ubuf[m][tid + 512] = (_Float16)(e2 * am - bm);
        if (tid < 232) ubuf[m][tid + 768] = (_Float16)(e3 * am - bm);
        if (tid < 24)  ubuf[m][NC + tid]  = (_Float16)0.0f;   // zero K padding 1000..1023
    }
    __syncthreads();

    // ---- wave 0: Gram = U * U^T via v_wmma_f32_16x16x32_f16, K = 1024 ----
    if (tid < 32) {
        const int lane = tid;
        int r = lane & 15;                 // A row / B column (clamped: rows >5 duplicate row 5,
        if (r > 5) r = 5;                  //  producing Gram entries we never read)
        const int kloA = (lane < 16) ? 0 : 8;   // 16-bit A layout: half-wave K interleave
        const int kloB = (lane < 16) ? 0 : 8;   // B layout: same half-wave K interleave as A

        v8f acc = {0.f, 0.f, 0.f, 0.f, 0.f, 0.f, 0.f, 0.f};

#pragma unroll 8
        for (int kk = 0; kk < KPAD; kk += 32) {
            // A operand: two contiguous 8x f16 (16B) runs -> ds_load_b128 each
            v8h alo = *(const v8h*)&ubuf[r][kk + kloA];
            v8h ahi = *(const v8h*)&ubuf[r][kk + kloA + 16];
            // B operand: two contiguous 8x f16 (16B) runs, same K order as A
            v8h blo = *(const v8h*)&ubuf[r][kk + kloB];
            v8h bhi = *(const v8h*)&ubuf[r][kk + kloB + 16];
            v16h A, Bm;
#pragma unroll
            for (int e = 0; e < 8; ++e) {
                A[e] = alo[e];  A[e + 8]  = ahi[e];
                Bm[e] = blo[e]; Bm[e + 8] = bhi[e];
            }
            acc = __builtin_amdgcn_wmma_f32_16x16x32_f16(
                false, A, false, Bm, (short)0, acc, false, false);
        }

        // C/D layout: VGPR j holds M=j (lanes 0-15) and M=j+8 (lanes 16-31), N = lane%16.
        // Strict upper triangle m<n, m,n<6 lives in lanes<16, VGPRs 0..4.
        float loc = 0.0f;
        if (lane < 16) {
            const int N = lane;
#pragma unroll
            for (int j = 0; j < 5; ++j)
                if (j < N && N < 6) loc += acc[j];
        }
        loc = wred_sum(loc);
        if (lane == 0) partial[b] = loc;   // = d_b (pre-scale)
    }
}

// Fixed-order deterministic final reduction: mean over B, times SCALE.
__global__ __launch_bounds__(256) void div6_reduce(const float* __restrict__ partial,
                                                   float* __restrict__ out)
{
    __shared__ float red[8];
    float a = 0.0f;
    for (int i = threadIdx.x; i < NB; i += 256) a += partial[i];
    a = bred_sum(a, red);
    if (threadIdx.x == 0) out[0] = a * (SCALE_F / (float)NB);
}

extern "C" void kernel_launch(void* const* d_in, const int* in_sizes, int n_in,
                              void* d_out, int out_size, void* d_ws, size_t ws_size,
                              hipStream_t stream) {
    (void)in_sizes; (void)n_in; (void)out_size; (void)ws_size;
    const float* o0 = (const float*)d_in[0];
    const float* o1 = (const float*)d_in[1];
    const float* o2 = (const float*)d_in[2];
    const float* o3 = (const float*)d_in[3];
    const float* o4 = (const float*)d_in[4];
    const float* o5 = (const float*)d_in[5];
    // d_in[6] (targets) is unused by the reference.
    float* partial = (float*)d_ws;          // NB floats
    float* out     = (float*)d_out;

    div6_main<<<NB, 256, 0, stream>>>(o0, o1, o2, o3, o4, o5, partial);
    div6_reduce<<<1, 256, 0, stream>>>(partial, out);
}
